// vector_attention_4123168604644
// MI455X (gfx1250) — matrix operand order repaired and checked by reading, not compiled, not run on hardware
//
#include <hip/hip_runtime.h>

// Problem constants (from reference): S=2048, B=32, D=256, H=4
#define S_LEN   2048
#define B_LEN   32
#define D_LEN   256
#define H_LEN   4
#define NROWS   (S_LEN * B_LEN)      // 65536 rows for [S,B,D] flattened as [N,D]
#define BD      (B_LEN * D_LEN)      // 8192
#define EPSV    1e-5f

typedef __attribute__((ext_vector_type(16))) __bf16 v16bf;
typedef __attribute__((ext_vector_type(8)))  __bf16 v8bf;
typedef __attribute__((ext_vector_type(8)))  float  v8f;

__device__ __forceinline__ unsigned short f2bf(float f) {
  unsigned int u = __builtin_bit_cast(unsigned int, f);
  unsigned int r = (u + 0x7FFFu + ((u >> 16) & 1u)) >> 16;  // round-to-nearest-even
  return (unsigned short)r;
}
__device__ __forceinline__ float bf2f(unsigned short h) {
  unsigned int u = ((unsigned int)h) << 16;
  return __builtin_bit_cast(float, u);
}

// CDNA5 async global->LDS copy (ASYNCcnt-tracked), 16 bytes per lane.
// GV addressing: 64-bit global address in a VGPR pair, LDS byte offset in VDST.
__device__ __forceinline__ void async_copy16(unsigned lds_off, const void* gp) {
  asm volatile("global_load_async_to_lds_b128 %0, %1, off"
               :: "v"(lds_off), "v"((unsigned long long)(size_t)gp)
               : "memory");
}
__device__ __forceinline__ void wait_async0() {
  asm volatile("s_wait_asynccnt 0x0" ::: "memory");
}

// ---------------------------------------------------------------------------
// f32 -> bf16 conversion (grid-stride)
// ---------------------------------------------------------------------------
__global__ void __launch_bounds__(256)
cvt_f32_bf16(const float* __restrict__ x, unsigned short* __restrict__ y, int n) {
  int i = blockIdx.x * blockDim.x + threadIdx.x;
  int stride = gridDim.x * blockDim.x;
  for (; i < n; i += stride) y[i] = f2bf(x[i]);
}

// ---------------------------------------------------------------------------
// WMMA GEMM:  Y[r][c] = sum_k X[r][k] * W[c][k] + bias[c]   (Y = X * W^T + b)
// X: [N][K] bf16 row-major.  W: [256][K] bf16 row-major.  Output width = 256.
// Block = 256 threads (8 waves); block covers 64 rows x 256 cols.
// Wave w covers cols [w*32, w*32+32) via 4 row-tiles x 2 col-tiles of 16x16.
//
// A tile (64x32 bf16 = 4KB) is staged into LDS with async copies (ASYNCcnt),
// double-buffered so the DMA of step k+1 overlaps the WMMAs of step k; this
// removes the 8x per-wave redundancy of direct A loads. B fragments are loaded
// straight from global (unique cols per wave; weight matrix is L2-resident).
// Fragment reads use contiguous k order (k = 16*(lane/16) + s) for both A(16x32) and B(32x16).
// In-place (Y == X) is safe: a block only writes rows it alone reads, and the
// in-loop s_wait_asynccnt + barrier complete all reads before the store phase.
// ---------------------------------------------------------------------------
__global__ void __launch_bounds__(256)
gemm_bias_wmma(const unsigned short* __restrict__ X,
               const unsigned short* __restrict__ W,
               const float* __restrict__ bias,
               unsigned short* __restrict__ Yb,   // bf16 out (nullable)
               float* __restrict__ Yf,            // f32 out (nullable)
               int N, int K, int relu) {
  __shared__ __attribute__((aligned(32))) unsigned short Atile[2][64 * 32];

  const int tid   = threadIdx.x;
  const int lane  = tid & 31;
  const int wave  = tid >> 5;
  const int rbase = blockIdx.x * 64;
  const int wcol  = wave * 32;
  const int lhalf = (lane < 16) ? 0 : 1;
  const int lmod  = lane & 15;

  // async-copy mapping: thread t moves 16B; row = t>>2 (0..63), chunk = t&3
  int arow = rbase + (tid >> 2);
  arow = (arow < N) ? arow : (N - 1);
  const int achk = tid & 3;
  const unsigned lds0 = (unsigned)(size_t)(&Atile[0][0]) + (unsigned)tid * 16u;

  v8f acc[4][2];
  const v8f zv = {0.f, 0.f, 0.f, 0.f, 0.f, 0.f, 0.f, 0.f};
#pragma unroll
  for (int rt = 0; rt < 4; ++rt)
#pragma unroll
    for (int ct = 0; ct < 2; ++ct) acc[rt][ct] = zv;

  // prologue: stage first A tile
  async_copy16(lds0, X + (size_t)arow * K + 0 + achk * 8);
  wait_async0();
  __syncthreads();

  int buf = 0;
  for (int k0 = 0; k0 < K; k0 += 32, buf ^= 1) {
    // kick off DMA of the next A tile into the other buffer
    if (k0 + 32 < K)
      async_copy16(lds0 + (unsigned)(buf ^ 1) * 4096u,
                   X + (size_t)arow * K + (k0 + 32) + achk * 8);

    // B fragments (K x 16): lane holds col = wcol+ct*16+lmod,
    // K = k0 + 16*lhalf .. +15 -> one contiguous 32B run of W row `col`.
    v16bf bfrag[2];
#pragma unroll
    for (int ct = 0; ct < 2; ++ct) {
      const int col = wcol + ct * 16 + lmod;
      const v8bf* p = (const v8bf*)(W + (size_t)col * K + k0 + 16 * lhalf);
      v8bf lo = p[0];
      v8bf hi = p[1];
#pragma unroll
      for (int e = 0; e < 8; ++e) { bfrag[ct][e] = lo[e]; bfrag[ct][8 + e] = hi[e]; }
    }

    // A fragments from LDS (row-major [64][32] bf16):
    // lanes 0-15: K=+0..15; lanes 16-31: K=+16..31.
    const unsigned short* Ab = &Atile[buf][0];
#pragma unroll
    for (int rt = 0; rt < 4; ++rt) {
      const int r = rt * 16 + lmod;
      v8bf lo = *(const v8bf*)(Ab + r * 32 + 16 * lhalf);
      v8bf hi = *(const v8bf*)(Ab + r * 32 + 16 * lhalf + 8);
      v16bf afrag;
#pragma unroll
      for (int e = 0; e < 8; ++e) { afrag[e] = lo[e]; afrag[8 + e] = hi[e]; }
#pragma unroll
      for (int ct = 0; ct < 2; ++ct) {
        acc[rt][ct] = __builtin_amdgcn_wmma_f32_16x16x32_bf16(
            false, afrag, false, bfrag[ct], (short)0, acc[rt][ct], false, false);
      }
    }

    wait_async0();     // next buffer is resident
    __syncthreads();   // everyone done reading current buffer + next one ready
  }

  // D layout: lane L, VGPR r -> M = r + 8*(L/16), Ncol = L%16
#pragma unroll
  for (int rt = 0; rt < 4; ++rt) {
#pragma unroll
    for (int ct = 0; ct < 2; ++ct) {
      const int col = wcol + ct * 16 + lmod;
      const float bcol = bias ? bias[col] : 0.f;
#pragma unroll
      for (int r = 0; r < 8; ++r) {
        const int row = rbase + rt * 16 + r + 8 * lhalf;
        if (row < N) {
          float val = acc[rt][ct][r] + bcol;
          if (relu) val = fmaxf(val, 0.f);
          if (Yb) Yb[(size_t)row * 256 + col] = f2bf(val);
          if (Yf) Yf[(size_t)row * 256 + col] = val;
        }
      }
    }
  }
}

// ---------------------------------------------------------------------------
// w = k - q (elementwise bf16)
// ---------------------------------------------------------------------------
__global__ void __launch_bounds__(256)
sub_bf16(const unsigned short* __restrict__ k, const unsigned short* __restrict__ q,
         unsigned short* __restrict__ w, int n) {
  int i = blockIdx.x * blockDim.x + threadIdx.x;
  int stride = gridDim.x * blockDim.x;
  for (; i < n; i += stride) w[i] = f2bf(bf2f(k[i]) - bf2f(q[i]));
}

// ---------------------------------------------------------------------------
// Per-channel partial sums (deterministic two-stage BN stats).
// Block b, thread c sums rows [b*RPB, (b+1)*RPB) of channel c.
// ---------------------------------------------------------------------------
__global__ void __launch_bounds__(256)
channel_stats_partial(const unsigned short* __restrict__ x,
                      float* __restrict__ psum, float* __restrict__ psumsq,
                      int rows_per_block) {
  const int c = threadIdx.x;
  const size_t r0 = (size_t)blockIdx.x * rows_per_block;
  float s = 0.f, ss = 0.f;
  for (int r = 0; r < rows_per_block; ++r) {
    float v = bf2f(x[(r0 + r) * 256 + c]);
    s += v; ss += v * v;
  }
  psum[blockIdx.x * 256 + c]   = s;
  psumsq[blockIdx.x * 256 + c] = ss;
}

__global__ void __launch_bounds__(256)
bn_finalize(const float* __restrict__ psum, const float* __restrict__ psumsq,
            const float* __restrict__ gamma, const float* __restrict__ beta,
            float* __restrict__ a, float* __restrict__ b, int nblk, float invN) {
  const int c = threadIdx.x;
  float s = 0.f, ss = 0.f;
  for (int i = 0; i < nblk; ++i) { s += psum[i * 256 + c]; ss += psumsq[i * 256 + c]; }
  const float m   = s * invN;
  const float var = ss * invN - m * m;
  const float sc  = gamma[c] * rsqrtf(var + EPSV);
  a[c] = sc;
  b[c] = beta[c] - m * sc;
}

__global__ void __launch_bounds__(256)
bn_relu_apply(unsigned short* __restrict__ w, const float* __restrict__ a,
              const float* __restrict__ b, int n) {
  int i = blockIdx.x * blockDim.x + threadIdx.x;
  int stride = gridDim.x * blockDim.x;
  for (; i < n; i += stride) {
    const int c = i & 255;
    float v = bf2f(w[i]) * a[c] + b[c];
    w[i] = f2bf(fmaxf(v, 0.f));
  }
}

// ---------------------------------------------------------------------------
// softmax over S (axis 0) fused with sum_S(softmax(w) * v) -> xcat[b, head*256+d]
// Block owns 64 consecutive (b,d) columns; threads = (bx = t&63, sy = t>>6).
// Row segments of 64 columns are contiguous 128B -> fully coalesced loads.
// Cross-sy reduction via LDS in fixed order (deterministic).
// ---------------------------------------------------------------------------
__global__ void __launch_bounds__(256)
softmax_wsum(const unsigned short* __restrict__ w, const unsigned short* __restrict__ v,
             float* __restrict__ xcat, int head) {
  __shared__ float red[4][64];
  const int t  = threadIdx.x;
  const int bx = t & 63;
  const int sy = t >> 6;
  const int bd = blockIdx.x * 64 + bx;

  // pass 1: max over S
  float m = -3.4e38f;
  for (int s = sy; s < S_LEN; s += 4) m = fmaxf(m, bf2f(w[(size_t)s * BD + bd]));
  red[sy][bx] = m;
  __syncthreads();
  m = fmaxf(fmaxf(red[0][bx], red[1][bx]), fmaxf(red[2][bx], red[3][bx]));
  __syncthreads();

  // pass 2: sum of exp
  float e = 0.f;
  for (int s = sy; s < S_LEN; s += 4) e += __expf(bf2f(w[(size_t)s * BD + bd]) - m);
  red[sy][bx] = e;
  __syncthreads();
  const float inv = 1.0f / (red[0][bx] + red[1][bx] + red[2][bx] + red[3][bx]);
  __syncthreads();

  // pass 3: weighted sum against v
  float acc = 0.f;
  for (int s = sy; s < S_LEN; s += 4)
    acc += __expf(bf2f(w[(size_t)s * BD + bd]) - m) * inv * bf2f(v[(size_t)s * BD + bd]);
  red[sy][bx] = acc;
  __syncthreads();
  if (sy == 0) {
    const float r = red[0][bx] + red[1][bx] + red[2][bx] + red[3][bx];
    const int b_ = bd >> 8, d_ = bd & 255;
    xcat[b_ * (H_LEN * D_LEN) + head * D_LEN + d_] = r;
  }
}

// ---------------------------------------------------------------------------
// Host-side launch
// ---------------------------------------------------------------------------
extern "C" void kernel_launch(void* const* d_in, const int* in_sizes, int n_in,
                              void* d_out, int out_size, void* d_ws, size_t ws_size,
                              hipStream_t stream) {
  (void)in_sizes; (void)n_in; (void)out_size; (void)ws_size;

  const float* q_in = (const float*)d_in[0];
  const float* k_in = (const float*)d_in[1];
  const float* v_in = (const float*)d_in[2];
  const float* wq  = (const float*)d_in[3];  const float* bq  = (const float*)d_in[4];
  const float* wk  = (const float*)d_in[5];  const float* bk  = (const float*)d_in[6];
  const float* wv  = (const float*)d_in[7];  const float* bv  = (const float*)d_in[8];
  const float* g1  = (const float*)d_in[9];  const float* be1 = (const float*)d_in[10];
  const float* wl1 = (const float*)d_in[11]; const float* bl1 = (const float*)d_in[12];
  const float* g2  = (const float*)d_in[13]; const float* be2 = (const float*)d_in[14];
  const float* wl2 = (const float*)d_in[15]; const float* bl2 = (const float*)d_in[16];
  const float* mw0 = (const float*)d_in[17]; const float* mb0 = (const float*)d_in[18];
  const float* mw1 = (const float*)d_in[19]; const float* mb1 = (const float*)d_in[20];
  const float* mw2 = (const float*)d_in[21]; const float* mb2 = (const float*)d_in[22];

  float* out = (float*)d_out;

  // ---- workspace carve-up (256B aligned chunks) ----
  char* ws = (char*)d_ws;
  size_t off = 0;
  auto alloc = [&](size_t bytes) -> char* {
    char* p = ws + off;
    off = (off + bytes + 255) & ~(size_t)255;
    return p;
  };
  const size_t NACT = (size_t)NROWS * 256;      // activation elements
  const size_t WHD  = (size_t)H_LEN * 256 * 256;

  unsigned short* qb   = (unsigned short*)alloc(NACT * 2);         // 32 MB
  unsigned short* kb   = (unsigned short*)alloc(NACT * 2);         // 32 MB
  unsigned short* vb   = (unsigned short*)alloc(NACT * 2);         // 32 MB
  unsigned short* wb   = (unsigned short*)alloc(NACT * 2);         // 32 MB
  unsigned short* wqb  = (unsigned short*)alloc(WHD * 2);
  unsigned short* wkb  = (unsigned short*)alloc(WHD * 2);
  unsigned short* wvb  = (unsigned short*)alloc(WHD * 2);
  unsigned short* wl1b = (unsigned short*)alloc(WHD * 2);
  unsigned short* wl2b = (unsigned short*)alloc(WHD * 2);
  unsigned short* mw0b = (unsigned short*)alloc((size_t)256 * 1024 * 2);
  unsigned short* mw1b = (unsigned short*)alloc((size_t)256 * 256 * 2);
  unsigned short* mw2b = (unsigned short*)alloc((size_t)256 * 256 * 2);
  float* psum   = (float*)alloc(256 * 256 * 4);
  float* psumsq = (float*)alloc(256 * 256 * 4);
  float* bn_a   = (float*)alloc(256 * 4);
  float* bn_b   = (float*)alloc(256 * 4);
  float* xcat   = (float*)alloc((size_t)B_LEN * 1024 * 4);
  unsigned short* xcatb = (unsigned short*)alloc((size_t)B_LEN * 1024 * 2);
  unsigned short* xm1b  = (unsigned short*)alloc((size_t)B_LEN * 256 * 2);
  unsigned short* xm2b  = (unsigned short*)alloc((size_t)B_LEN * 256 * 2);

  const int NEL = (int)NACT;                    // 16,777,216
  const dim3 blk(256);
  const dim3 gridBig(16384);
  const dim3 gridGemm(NROWS / 64);              // 1024
  const dim3 gridStats(256);                    // 256 rows each
  const dim3 gridSm(BD / 64);                   // 128 blocks x 64 columns

  // ---- 1. convert activations and weights to bf16 ----
  cvt_f32_bf16<<<gridBig, blk, 0, stream>>>(q_in, qb, NEL);
  cvt_f32_bf16<<<gridBig, blk, 0, stream>>>(k_in, kb, NEL);
  cvt_f32_bf16<<<gridBig, blk, 0, stream>>>(v_in, vb, NEL);
  cvt_f32_bf16<<<dim3(1024), blk, 0, stream>>>(wq,  wqb,  (int)WHD);
  cvt_f32_bf16<<<dim3(1024), blk, 0, stream>>>(wk,  wkb,  (int)WHD);
  cvt_f32_bf16<<<dim3(1024), blk, 0, stream>>>(wv,  wvb,  (int)WHD);
  cvt_f32_bf16<<<dim3(1024), blk, 0, stream>>>(wl1, wl1b, (int)WHD);
  cvt_f32_bf16<<<dim3(1024), blk, 0, stream>>>(wl2, wl2b, (int)WHD);
  cvt_f32_bf16<<<dim3(1024), blk, 0, stream>>>(mw0, mw0b, 256 * 1024);
  cvt_f32_bf16<<<dim3(256),  blk, 0, stream>>>(mw1, mw1b, 256 * 256);
  cvt_f32_bf16<<<dim3(256),  blk, 0, stream>>>(mw2, mw2b, 256 * 256);

  // ---- 2. per-head pipeline (q/k/v projections chain across heads) ----
  for (int h = 0; h < H_LEN; ++h) {
    const size_t wo = (size_t)h * 256 * 256;
    const int bo = h * 256;
    // chained projections, in-place
    gemm_bias_wmma<<<gridGemm, blk, 0, stream>>>(qb, wqb + wo, bq + bo, qb, nullptr, NROWS, 256, 0);
    gemm_bias_wmma<<<gridGemm, blk, 0, stream>>>(kb, wkb + wo, bk + bo, kb, nullptr, NROWS, 256, 0);
    gemm_bias_wmma<<<gridGemm, blk, 0, stream>>>(vb, wvb + wo, bv + bo, vb, nullptr, NROWS, 256, 0);
    // w = k - q
    sub_bf16<<<gridBig, blk, 0, stream>>>(kb, qb, wb, NEL);
    // BN1 -> ReLU
    channel_stats_partial<<<gridStats, blk, 0, stream>>>(wb, psum, psumsq, NROWS / 256);
    bn_finalize<<<dim3(1), blk, 0, stream>>>(psum, psumsq, g1 + bo, be1 + bo, bn_a, bn_b,
                                             256, 1.0f / (float)NROWS);
    bn_relu_apply<<<gridBig, blk, 0, stream>>>(wb, bn_a, bn_b, NEL);
    // Linear 1 (in-place)
    gemm_bias_wmma<<<gridGemm, blk, 0, stream>>>(wb, wl1b + wo, bl1 + bo, wb, nullptr, NROWS, 256, 0);
    // BN2 -> ReLU
    channel_stats_partial<<<gridStats, blk, 0, stream>>>(wb, psum, psumsq, NROWS / 256);
    bn_finalize<<<dim3(1), blk, 0, stream>>>(psum, psumsq, g2 + bo, be2 + bo, bn_a, bn_b,
                                             256, 1.0f / (float)NROWS);
    bn_relu_apply<<<gridBig, blk, 0, stream>>>(wb, bn_a, bn_b, NEL);
    // Linear 2 (in-place)
    gemm_bias_wmma<<<gridGemm, blk, 0, stream>>>(wb, wl2b + wo, bl2 + bo, wb, nullptr, NROWS, 256, 0);
    // softmax over S + weighted sum against v -> xcat[:, h*256:(h+1)*256]
    softmax_wsum<<<gridSm, blk, 0, stream>>>(wb, vb, xcat, h);
  }

  // ---- 3. final MLP on [B=32, 1024] ----
  cvt_f32_bf16<<<dim3(128), blk, 0, stream>>>(xcat, xcatb, B_LEN * 1024);
  gemm_bias_wmma<<<dim3(1), blk, 0, stream>>>(xcatb, mw0b, mb0, xm1b, nullptr, B_LEN, 1024, 1);
  gemm_bias_wmma<<<dim3(1), blk, 0, stream>>>(xm1b,  mw1b, mb1, xm2b, nullptr, B_LEN, 256, 1);
  gemm_bias_wmma<<<dim3(1), blk, 0, stream>>>(xm2b,  mw2b, mb2, nullptr, out, B_LEN, 256, 0);
}
